// FCOS_70463233459029
// MI455X (gfx1250) — matrix operand order repaired and checked by reading, not compiled, not run on hardware
//
#include <hip/hip_runtime.h>
#include <hip/hip_bf16.h>

typedef _Float16 f16;
typedef __attribute__((ext_vector_type(16))) _Float16 v16h;
typedef __attribute__((ext_vector_type(8)))  float    v8f;
typedef __attribute__((ext_vector_type(4)))  int      v4i;

#define KTOT   2304     // 9 taps * 256 in-channels
#define NKCHUNK 72      // KTOT / 32
#define TILE_M 128
#define TILE_N 64
#define N_BATCH 2
#define C_FPN  256
#define TOT_LOC 20267

#if __has_builtin(__builtin_amdgcn_global_load_async_to_lds_b128) && \
    __has_builtin(__builtin_amdgcn_s_wait_asynccnt)
#define USE_ASYNC 1
#else
#define USE_ASYNC 0
#endif

union FragH { v16h h; uint4 u[2]; };

#if USE_ASYNC
typedef __attribute__((address_space(1))) v4i gv4i;   // global int4
typedef __attribute__((address_space(3))) v4i lv4i;   // LDS int4
#endif

__device__ __forceinline__ void stage_b128(const f16* g, f16* l)
{
#if USE_ASYNC
    __builtin_amdgcn_global_load_async_to_lds_b128((gv4i*)g, (lv4i*)l, 0, 0);
#else
    *(uint4*)l = *(const uint4*)g;
#endif
}

// ---------------------------------------------------------------------------
// Implicit-GEMM 3x3 conv (stride 1, SAME) via WMMA f32_16x16x32_f16.
//   X  : f16 activations, HWC [n][H*W][256]
//   Wp : f16 packed weights, [o][k], k = tap*256 + c, rows padded to mult of 64
//   zb : >=128B of zeroed global memory (source for OOB/border lanes)
//   Out: fp32 at Out + n*nStrideOut + m*ldo + colOff + o
// 256 threads = 8 waves; block tile 128(M) x 64(N); wave tile 16(M) x 64(N).
// Triple-buffered LDS, async global->LDS staging 2 chunks ahead, one barrier
// per K-step:  wait(own chunk-k asyncs) -> barrier -> issue(k+2) -> compute(k)
// ---------------------------------------------------------------------------
__global__ __launch_bounds__(256)
void fcos_conv_gemm(const f16* __restrict__ X, const f16* __restrict__ Wp,
                    const float* __restrict__ bias, const f16* __restrict__ zb,
                    float* __restrict__ Out,
                    int H, int W, int HW, int Ncols, int ldo, int colOff,
                    unsigned relumask, size_t nStrideX, size_t nStrideOut)
{
    __shared__ __align__(16) f16 As[3][TILE_M * 32];
    __shared__ __align__(16) f16 Bs[3][TILE_N * 32];

    const int tid  = threadIdx.x;
    const int wave = tid >> 5;
    const int lane = tid & 31;
    const int ln   = lane & 15;
    const int hf   = lane >> 4;
    const int n    = blockIdx.z;
    const int m0   = blockIdx.x * TILE_M;
    const int n0   = blockIdx.y * TILE_N;
    const f16* Xn  = X + (size_t)n * nStrideX;

    // --- staging roles ---
    const int ar = tid >> 1, ah = tid & 1;   // A: row 0..127, 32B half
    const int br = tid >> 2, bq = tid & 3;   // B: row 0..63,  16B quarter
    const int am = m0 + ar;
    const bool am_ok = (am < HW);
    const int ay = am_ok ? (am / W) : 0;
    const int ax = am_ok ? (am % W) : 0;
    const f16* bgp = Wp + (size_t)(n0 + br) * KTOT + bq * 8;  // rows are padded

    // issue-side A state (recomputed at each tap boundary)
    const f16* apix = Xn;
    bool aok = false;

    auto issue = [&](int ik, int buf) {
        if ((ik & 7) == 0) {
            const int tap = ik >> 3;
            const int ty  = tap / 3;
            const int dy  = ty - 1, dx = (tap - ty * 3) - 1;
            const int yy  = ay + dy, xx = ax + dx;
            aok  = am_ok && yy >= 0 && yy < H && xx >= 0 && xx < W;
            apix = Xn + (size_t)(yy * W + xx) * C_FPN;
        }
        // A: 32B per thread (2 x b128); border/tail lanes pull from zero page
        f16* la = &As[buf][ar * 32 + ah * 16];
        const f16* ga = aok ? (apix + ((ik & 7) * 32 + ah * 16)) : zb;
        stage_b128(ga,     la);
        stage_b128(ga + 8, la + 8);
        // B: 16B per thread (1 x b128); weight rows zero-padded, always valid
        stage_b128(bgp + ik * 32, &Bs[buf][br * 32 + bq * 8]);
    };

    v8f acc[4] = {v8f{0.f,0.f,0.f,0.f,0.f,0.f,0.f,0.f},
                  v8f{0.f,0.f,0.f,0.f,0.f,0.f,0.f,0.f},
                  v8f{0.f,0.f,0.f,0.f,0.f,0.f,0.f,0.f},
                  v8f{0.f,0.f,0.f,0.f,0.f,0.f,0.f,0.f}};

    issue(0, 0);
    issue(1, 1);
    int cur = 0;
    for (int kt = 0; kt < NKCHUNK; ++kt) {
        // 1) my chunk-kt asyncs done (only chunk kt+1 may remain in flight)
#if USE_ASYNC
        if (kt + 1 < NKCHUNK) __builtin_amdgcn_s_wait_asynccnt(3);
        else                  __builtin_amdgcn_s_wait_asynccnt(0);
#endif
        // 2) all waves: chunk-kt data in LDS, buffer kt-1 fully consumed
        __syncthreads();
        // 3) prefetch chunk kt+2 into the buffer freed at step kt-1
        if (kt + 2 < NKCHUNK) {
            int ib = cur + 2; if (ib >= 3) ib -= 3;
            issue(kt + 2, ib);
        }

        // 4) compute chunk kt
        // A fragment (16x32 f16): lane = row M=ln; K = 16h..16h+15
        FragH fa;
        {
            const f16* arow = &As[cur][(wave * 16 + ln) * 32];
            fa.u[0] = *(const uint4*)(arow + hf * 16);
            fa.u[1] = *(const uint4*)(arow + hf * 16 + 8);
        }
        // B fragments (32x16 f16): lane = col N=ln; K = 16h..16h+15
        FragH fb[4];
        #pragma unroll
        for (int j = 0; j < 4; ++j) {
            const f16* brow = &Bs[cur][(j * 16 + ln) * 32];
            fb[j].u[0] = *(const uint4*)(brow + hf * 16);
            fb[j].u[1] = *(const uint4*)(brow + hf * 16 + 8);
        }
        #pragma unroll
        for (int j = 0; j < 4; ++j) {
            acc[j] = __builtin_amdgcn_wmma_f32_16x16x32_f16(
                         false, fa.h, false, fb[j].h, (short)0, acc[j], false, false);
        }

        cur = (cur == 2) ? 0 : cur + 1;
    }

    // C layout: lane ln = N col; VGPR r -> M = r + 8*hf
    const int mw = m0 + wave * 16;
    #pragma unroll
    for (int j = 0; j < 4; ++j) {
        #pragma unroll
        for (int r = 0; r < 8; ++r) {
            const int mm = mw + r + 8 * hf;
            const int nn = n0 + j * 16 + ln;
            if (mm < HW && nn < Ncols) {
                float v = acc[j][r] + bias[nn];
                if ((relumask >> (nn & 31)) & 1u) v = fmaxf(v, 0.f);
                Out[(size_t)n * nStrideOut + (size_t)mm * ldo + colOff + nn] = v;
            }
        }
    }
}

// ---------------------------------------------------------------------------
// GroupNorm stats: one block per (n, group); Y is fp32 HWC [n][HW][256].
// ---------------------------------------------------------------------------
__global__ __launch_bounds__(256)
void fcos_gn_stats(const float* __restrict__ Y, float2* __restrict__ stats, int HW)
{
    const int b = blockIdx.x;           // n*16 + g
    const int nn = b >> 4, g = b & 15;
    const float* Yn = Y + (size_t)nn * HW * C_FPN;
    const int t = threadIdx.x;
    const int c = g * 16 + (t & 15);
    float s = 0.f, ss = 0.f;
    for (int p = t >> 4; p < HW; p += 16) {
        float v = Yn[(size_t)p * C_FPN + c];
        s += v; ss += v * v;
    }
    __shared__ float sh[256], sh2[256];
    sh[t] = s; sh2[t] = ss;
    __syncthreads();
    for (int off = 128; off > 0; off >>= 1) {
        if (t < off) { sh[t] += sh[t + off]; sh2[t] += sh2[t + off]; }
        __syncthreads();
    }
    if (t == 0) {
        const float cnt = (float)HW * 16.f;
        const float mu  = sh[0] / cnt;
        const float var = sh2[0] / cnt - mu * mu;
        stats[b] = make_float2(mu, rsqrtf(var + 1e-5f));
    }
}

// normalize + affine + ReLU + convert to f16 (next conv's input)
__global__ __launch_bounds__(256)
void fcos_gn_norm(const float* __restrict__ Y, const float2* __restrict__ stats,
                  const float* __restrict__ gs, const float* __restrict__ gb,
                  f16* __restrict__ Xo, int HW)
{
    const size_t total = (size_t)N_BATCH * HW * C_FPN;
    size_t i = (size_t)blockIdx.x * 256 + threadIdx.x;
    if (i >= total) return;
    const int c = (int)(i & (C_FPN - 1));
    const int nn = (int)(i / ((size_t)HW * C_FPN));
    const float2 st = stats[nn * 16 + (c >> 4)];
    float v = (Y[i] - st.x) * st.y * gs[c] + gb[c];
    v = fmaxf(v, 0.f);
    Xo[i] = (f16)v;
}

// NCHW fp32 feature -> HWC f16
__global__ __launch_bounds__(256)
void fcos_pack_x(const float* __restrict__ in, f16* __restrict__ out, int HW)
{
    const size_t total = (size_t)N_BATCH * HW * C_FPN;
    size_t i = (size_t)blockIdx.x * 256 + threadIdx.x;
    if (i >= total) return;
    const int c   = (int)(i & (C_FPN - 1));
    const size_t rem = i >> 8;            // n*HW + p
    const int nn  = (int)(rem / HW);
    const int p   = (int)(rem - (size_t)nn * HW);
    out[i] = (f16)in[((size_t)nn * C_FPN + c) * HW + p];
}

// [O][256][3][3] fp32 -> packed f16 [O][k = tap*256 + c]
__global__ __launch_bounds__(256)
void fcos_pack_w(const float* __restrict__ w, f16* __restrict__ out, int O)
{
    const size_t total = (size_t)O * KTOT;
    size_t i = (size_t)blockIdx.x * 256 + threadIdx.x;
    if (i >= total) return;
    const int k = (int)(i % KTOT);
    const int o = (int)(i / KTOT);
    const int t = k >> 8, c = k & 255;
    out[i] = (f16)w[((size_t)o * C_FPN + c) * 9 + t];
}

__global__ void fcos_zero_h(f16* p, int count)
{
    int i = blockIdx.x * 256 + threadIdx.x;
    if (i < count) p[i] = (f16)0.f;
}

__global__ void fcos_make_reg_bias(const float* __restrict__ box_b,
                                   const float* __restrict__ ctr_b,
                                   float* __restrict__ out)
{
    int i = threadIdx.x;
    if (i < 16) out[i] = (i < 4) ? box_b[i] : ((i == 4) ? ctr_b[0] : 0.f);
}

// ---------------------------------------------------------------------------
extern "C" void kernel_launch(void* const* d_in, const int* in_sizes, int n_in,
                              void* d_out, int out_size, void* d_ws, size_t ws_size,
                              hipStream_t stream) {
    (void)in_sizes; (void)n_in; (void)out_size; (void)ws_size;

    static const int LH[5] = {100, 50, 25, 13, 7};
    static const int LW[5] = {152, 76, 38, 19, 10};
    static const int LOFF[5] = {0, 15200, 19000, 19950, 20197};
    const int HWMAX = 15200;

    const float* feats[5];
    for (int i = 0; i < 5; ++i) feats[i] = (const float*)d_in[i];
    const float* cls_conv_w = (const float*)d_in[5];
    const float* cls_conv_b = (const float*)d_in[6];
    const float* cls_gn_s   = (const float*)d_in[7];
    const float* cls_gn_b   = (const float*)d_in[8];
    const float* cls_out_w  = (const float*)d_in[9];
    const float* cls_out_b  = (const float*)d_in[10];
    const float* reg_conv_w = (const float*)d_in[11];
    const float* reg_conv_b = (const float*)d_in[12];
    const float* reg_gn_s   = (const float*)d_in[13];
    const float* reg_gn_b   = (const float*)d_in[14];
    const float* box_w      = (const float*)d_in[15];
    const float* box_b      = (const float*)d_in[16];
    const float* ctr_w      = (const float*)d_in[17];
    const float* ctr_b      = (const float*)d_in[18];
    float* out = (float*)d_out;

    // ---- workspace carve-up (256B aligned) ----
    char* ws = (char*)d_ws;
    size_t off = 0;
    auto carve = [&](size_t bytes) -> char* {
        off = (off + 255) & ~(size_t)255;
        char* p = ws + off;
        off += bytes;
        return p;
    };
    f16*    Wcls   = (f16*)   carve((size_t)3 * C_FPN * KTOT * sizeof(f16));
    f16*    Wreg   = (f16*)   carve((size_t)3 * C_FPN * KTOT * sizeof(f16));
    f16*    Wch    = (f16*)   carve((size_t)128 * KTOT * sizeof(f16)); // 80 -> pad 128
    f16*    Wrh    = (f16*)   carve((size_t)64  * KTOT * sizeof(f16)); // 5  -> pad 64
    float*  breg   = (float*) carve(16 * sizeof(float));
    f16*    zbuf   = (f16*)   carve(64 * sizeof(f16));                 // 128B zero page
    float2* stats  = (float2*)carve(N_BATCH * 16 * sizeof(float2));
    f16*    Xin    = (f16*)   carve((size_t)N_BATCH * HWMAX * C_FPN * sizeof(f16));
    f16*    Xt     = (f16*)   carve((size_t)N_BATCH * HWMAX * C_FPN * sizeof(f16));
    float*  Ybuf   = (float*) carve((size_t)N_BATCH * HWMAX * C_FPN * sizeof(float));

    auto cdiv = [](size_t a, size_t b) { return (unsigned)((a + b - 1) / b); };

    // ---- weight / bias packing (every call; deterministic) ----
    for (int i = 0; i < 3; ++i) {
        size_t wstride = (size_t)C_FPN * C_FPN * 9;
        fcos_pack_w<<<cdiv((size_t)C_FPN * KTOT, 256), 256, 0, stream>>>(
            cls_conv_w + i * wstride, Wcls + (size_t)i * C_FPN * KTOT, C_FPN);
        fcos_pack_w<<<cdiv((size_t)C_FPN * KTOT, 256), 256, 0, stream>>>(
            reg_conv_w + i * wstride, Wreg + (size_t)i * C_FPN * KTOT, C_FPN);
    }
    fcos_zero_h<<<cdiv((size_t)128 * KTOT, 256), 256, 0, stream>>>(Wch, 128 * KTOT);
    fcos_pack_w<<<cdiv((size_t)80 * KTOT, 256), 256, 0, stream>>>(cls_out_w, Wch, 80);
    fcos_zero_h<<<cdiv((size_t)64 * KTOT, 256), 256, 0, stream>>>(Wrh, 64 * KTOT);
    fcos_pack_w<<<cdiv((size_t)4 * KTOT, 256), 256, 0, stream>>>(box_w, Wrh, 4);
    fcos_pack_w<<<cdiv((size_t)1 * KTOT, 256), 256, 0, stream>>>(ctr_w, Wrh + (size_t)4 * KTOT, 1);
    fcos_zero_h<<<1, 64, 0, stream>>>(zbuf, 64);
    fcos_make_reg_bias<<<1, 16, 0, stream>>>(box_b, ctr_b, breg);

    // ---- per-level pipeline ----
    for (int l = 0; l < 5; ++l) {
        const int H = LH[l], W = LW[l], HW = H * W;
        const size_t nStrideX = (size_t)HW * C_FPN;
        const size_t elems = (size_t)N_BATCH * HW * C_FPN;

        fcos_pack_x<<<cdiv(elems, 256), 256, 0, stream>>>(feats[l], Xin, HW);

        for (int tower = 0; tower < 2; ++tower) {
            const f16*   Wt = tower ? Wreg : Wcls;
            const float* cb = tower ? reg_conv_b : cls_conv_b;
            const float* gs = tower ? reg_gn_s : cls_gn_s;
            const float* gb = tower ? reg_gn_b : cls_gn_b;

            const f16* src = Xin;
            for (int i = 0; i < 3; ++i) {
                dim3 grid(cdiv(HW, TILE_M), cdiv(C_FPN, TILE_N), N_BATCH);
                fcos_conv_gemm<<<grid, 256, 0, stream>>>(
                    src, Wt + (size_t)i * C_FPN * KTOT, cb + i * C_FPN, zbuf, Ybuf,
                    H, W, HW, C_FPN, C_FPN, 0, 0u, nStrideX, nStrideX);
                fcos_gn_stats<<<N_BATCH * 16, 256, 0, stream>>>(Ybuf, stats, HW);
                fcos_gn_norm<<<cdiv(elems, 256), 256, 0, stream>>>(
                    Ybuf, stats, gs + i * C_FPN, gb + i * C_FPN, Xt, HW);
                src = Xt;
            }

            float* outLvl = out + (size_t)LOFF[l] * 85;
            const size_t nStrideOut = (size_t)TOT_LOC * 85;
            if (tower == 0) {
                // cls head: 80 logits, cols [0,80)
                dim3 grid(cdiv(HW, TILE_M), 2, N_BATCH);
                fcos_conv_gemm<<<grid, 256, 0, stream>>>(
                    Xt, Wch, cls_out_b, zbuf, outLvl,
                    H, W, HW, 80, 85, 0, 0u, nStrideX, nStrideOut);
            } else {
                // reg head: 4 box (ReLU) + 1 ctr, cols [80,85)
                dim3 grid(cdiv(HW, TILE_M), 1, N_BATCH);
                fcos_conv_gemm<<<grid, 256, 0, stream>>>(
                    Xt, Wrh, breg, zbuf, outLvl,
                    H, W, HW, 5, 85, 80, 0xFu, nStrideX, nStrideOut);
            }
        }
    }
}
